// Encoder_75754633167557
// MI455X (gfx1250) — matrix operand order repaired and checked by reading, not compiled, not run on hardware
//
#include <hip/hip_runtime.h>
#include <hip/hip_bf16.h>
#include <math.h>

// ---------------- types ----------------
typedef __bf16 bf16_t;
typedef __attribute__((ext_vector_type(16))) __bf16 v16bf;
typedef __attribute__((ext_vector_type(8)))  __bf16 v8bf;
typedef __attribute__((ext_vector_type(8)))  float  v8f;
typedef __attribute__((ext_vector_type(4)))  unsigned int v4u;
typedef __attribute__((ext_vector_type(8)))  int v8i_t;
typedef __attribute__((ext_vector_type(4)))  int v4i_t;

#define B_SZ   64
#define T_SZ   512
#define M_TOT  (B_SZ * T_SZ)   // 32768
#define IN_DIM 800
#define HID    400
#define R4     1600
#define KP     416              // HID padded to multiple of 32 (WMMA K)
#define NCHUNK 10               // workgroups per direction in recurrence
#define RCHUNK 40               // hidden units per workgroup
#define NLOC   (4 * RCHUNK)     // 160 interleaved gate columns per WG

#if defined(__has_builtin)
#  if __has_builtin(__builtin_amdgcn_tensor_load_to_lds) && \
      __has_builtin(__builtin_amdgcn_s_wait_tensorcnt)
#    define USE_TDM 1
#  endif
#endif

// ---------------- workspace layout (all 256B aligned) ----------------
static constexpr size_t O_LW    = 0;                               // leftW bf16 [400][800]
static constexpr size_t O_RW    = O_LW    + 400 * 800 * 2;         // rightW bf16
static constexpr size_t O_WIHL  = O_RW    + 400 * 800 * 2;         // Wih_l bf16 [1600][416]
static constexpr size_t O_WIHR  = O_WIHL  + (size_t)R4 * KP * 2;
static constexpr size_t O_WHHL  = O_WIHR  + (size_t)R4 * KP * 2;   // Whh bf16 [1600][416]
static constexpr size_t O_WHHR  = O_WHHL  + (size_t)R4 * KP * 2;
static constexpr size_t O_BIASL = O_WHHR  + (size_t)R4 * KP * 2;   // bih+bhh f32 [1600]
static constexpr size_t O_BIASR = O_BIASL + R4 * 4;
static constexpr size_t O_HBUFL = O_BIASR + R4 * 4;                // h double buffer bf16 [2][64][416]
static constexpr size_t O_HBUFR = O_HBUFL + (size_t)2 * B_SZ * KP * 2;
static constexpr size_t O_CTR   = O_HBUFR + (size_t)2 * B_SZ * KP * 2; // 2 counters (+pad)
static constexpr size_t O_INL   = O_CTR   + 256;                   // leftIn bf16 [32768][416]
static constexpr size_t O_INR   = O_INL   + (size_t)M_TOT * KP * 2;
static constexpr size_t O_GXL   = O_INR   + (size_t)M_TOT * KP * 2; // gx f32 [T*B][1600]
static constexpr size_t O_GXR   = O_GXL   + (size_t)M_TOT * R4 * 4;
// total ~482 MB

// ---------------- helpers ----------------
__device__ __forceinline__ v16bf cat8(v8bf lo, v8bf hi) {
  return __builtin_shufflevector(lo, hi, 0,1,2,3,4,5,6,7,8,9,10,11,12,13,14,15);
}
__device__ __forceinline__ v8f wmma_bf16(v16bf a, v16bf b, v8f c) {
  // D = A(16x32) * B(32x16) + C, f32 accumulate
  return __builtin_amdgcn_wmma_f32_16x16x32_bf16(false, a, false, b, (short)0, c, false, false);
}
__device__ __forceinline__ float sigmoidf_fast(float x) {
  return 1.0f / (1.0f + __expf(-x));
}
// Branchless gather of one 8-wide chunk of the 800-dim concat row.
// Chunks are 8-aligned and never cross a 200-col segment (200 % 8 == 0).
__device__ __forceinline__ v8bf embed8(int k8, int ci, int bi,
    const float* charW, const float* extCharW,
    const float* bicharW, const float* extBiCharW) {
  const int  seg  = k8 / 200;                   // 0..3
  const int  col  = k8 - seg * 200;
  const bool isBi = (seg >= 2);
  const bool zro  = isBi && (bi < 0);           // shifted-out bigram row
  const int  row  = isBi ? (zro ? 0 : bi) : ci;
  const float* base = (seg == 0) ? charW
                    : (seg == 1) ? extCharW
                    : (seg == 2) ? bicharW
                    :              extBiCharW;  // pointer selects -> v_cndmask
  const float* p = base + (size_t)row * 200 + col;
  const float sc = zro ? 0.0f : 1.0f;
  float4 f0 = ((const float4*)p)[0];
  float4 f1 = ((const float4*)p)[1];
  v8bf r;
  r[0]=(bf16_t)(f0.x*sc); r[1]=(bf16_t)(f0.y*sc);
  r[2]=(bf16_t)(f0.z*sc); r[3]=(bf16_t)(f0.w*sc);
  r[4]=(bf16_t)(f1.x*sc); r[5]=(bf16_t)(f1.y*sc);
  r[6]=(bf16_t)(f1.z*sc); r[7]=(bf16_t)(f1.w*sc);
  return r;
}

// ---------------- prep kernels ----------------
__global__ void k_cvt_dense(const float* __restrict__ src, bf16_t* __restrict__ dst, int n) {
  int i = blockIdx.x * blockDim.x + threadIdx.x;
  if (i < n) dst[i] = (bf16_t)src[i];
}
__global__ void k_cvt_pad416(const float* __restrict__ src, bf16_t* __restrict__ dst) {
  int i = blockIdx.x * blockDim.x + threadIdx.x;        // over 1600*416
  if (i >= R4 * KP) return;
  int r = i / KP, k = i - r * KP;
  dst[i] = (k < HID) ? (bf16_t)src[r * HID + k] : (bf16_t)0.0f;
}
__global__ void k_bias_sum(const float* __restrict__ a, const float* __restrict__ b,
                           float* __restrict__ d) {
  int i = blockIdx.x * blockDim.x + threadIdx.x;
  if (i < R4) d[i] = a[i] + b[i];
}
__global__ void k_zero_pad_cols(bf16_t* __restrict__ In) {   // zero cols 400..415
  int i = blockIdx.x * blockDim.x + threadIdx.x;             // over 32768*16
  if (i < M_TOT * 16) In[(size_t)(i >> 4) * KP + HID + (i & 15)] = (bf16_t)0.0f;
}
__global__ void k_zero_u32(unsigned* __restrict__ p, int n) {
  int i = blockIdx.x * blockDim.x + threadIdx.x;
  if (i < n) p[i] = 0u;
}

// ---------------- projection: In = tanh(concat @ W^T + b), fused gather ----
// grid (256, 5, 2) block 256.  Wave w: rows [bx*128+w*16, +16), cols [by*80, +80)
__global__ __launch_bounds__(256) void k_proj(
    const int* __restrict__ charIdx, const int* __restrict__ bicharIdx,
    const float* __restrict__ charW, const float* __restrict__ extCharW,
    const float* __restrict__ bicharW, const float* __restrict__ extBiCharW,
    const bf16_t* __restrict__ Wbf_l, const bf16_t* __restrict__ Wbf_r,
    const float* __restrict__ bias_l, const float* __restrict__ bias_r,
    bf16_t* __restrict__ In_l, bf16_t* __restrict__ In_r) {
  const int dir = blockIdx.z;
  const bf16_t* Wbf = dir ? Wbf_r : Wbf_l;
  const float* bias = dir ? bias_r : bias_l;
  bf16_t* In = dir ? In_r : In_l;

  const int tid = threadIdx.x, w = tid >> 5, lane = tid & 31;
  const int mc = lane & 15, kg = lane >> 4;
  const int r0 = blockIdx.x * 128 + w * 16;
  const int n0 = blockIdx.y * 80;
  const int r = r0 + mc;
  const int t = r & (T_SZ - 1);
  const int ci = charIdx[r];
  int bi;
  if (dir == 0) bi = (t == T_SZ - 1) ? -1 : bicharIdx[r + 1];  // left: shifted bigram
  else          bi = bicharIdx[r];

  v8f acc[5];
#pragma unroll
  for (int i = 0; i < 5; ++i) acc[i] = {};

  for (int ks = 0; ks < 25; ++ks) {            // K = 800
    const int k0 = ks * 32;
    v8bf alo = embed8(k0 + 8 * kg,      ci, bi, charW, extCharW, bicharW, extBiCharW);
    v8bf ahi = embed8(k0 + 16 + 8 * kg, ci, bi, charW, extCharW, bicharW, extBiCharW);
    v16bf a = cat8(alo, ahi);
#pragma unroll
    for (int nt = 0; nt < 5; ++nt) {
      const bf16_t* bp = Wbf + (size_t)(n0 + nt * 16 + mc) * IN_DIM + k0 + 8 * kg;
      acc[nt] = wmma_bf16(a, cat8(*(const v8bf*)bp, *(const v8bf*)(bp + 16)), acc[nt]);
    }
  }
#pragma unroll
  for (int nt = 0; nt < 5; ++nt) {
    const int n = n0 + nt * 16 + mc;
    const float bv = bias[n];
#pragma unroll
    for (int v = 0; v < 8; ++v) {
      const int row = r0 + v + 8 * kg;
      In[(size_t)row * KP + n] = (bf16_t)tanhf(acc[nt][v] + bv);
    }
  }
}

// ---------------- x-side gates: gx[t*B+b][:] = In @ Wih^T + (bih+bhh) ------
// grid (256, 20, 2) block 256
__global__ __launch_bounds__(256) void k_gates(
    const bf16_t* __restrict__ In_l, const bf16_t* __restrict__ In_r,
    const bf16_t* __restrict__ Wih_l, const bf16_t* __restrict__ Wih_r,
    const float* __restrict__ bias_l, const float* __restrict__ bias_r,
    float* __restrict__ gx_l, float* __restrict__ gx_r) {
  const int dir = blockIdx.z;
  const bf16_t* In  = dir ? In_r  : In_l;
  const bf16_t* Wih = dir ? Wih_r : Wih_l;
  const float* bias = dir ? bias_r : bias_l;
  float* gx = dir ? gx_r : gx_l;

  const int tid = threadIdx.x, w = tid >> 5, lane = tid & 31;
  const int mc = lane & 15, kg = lane >> 4;
  const int r0 = blockIdx.x * 128 + w * 16;
  const int n0 = blockIdx.y * 80;
  const bf16_t* arow = In + (size_t)(r0 + mc) * KP;

  v8f acc[5];
#pragma unroll
  for (int i = 0; i < 5; ++i) acc[i] = {};

  for (int ks = 0; ks < 13; ++ks) {            // K = 416 (padded)
    const int k0 = ks * 32;
    v16bf a = cat8(*(const v8bf*)(arow + k0 + 8 * kg),
                   *(const v8bf*)(arow + k0 + 16 + 8 * kg));
#pragma unroll
    for (int nt = 0; nt < 5; ++nt) {
      const bf16_t* bp = Wih + (size_t)(n0 + nt * 16 + mc) * KP + k0 + 8 * kg;
      acc[nt] = wmma_bf16(a, cat8(*(const v8bf*)bp, *(const v8bf*)(bp + 16)), acc[nt]);
    }
  }
#pragma unroll
  for (int nt = 0; nt < 5; ++nt) {
    const int n = n0 + nt * 16 + mc;
    const float bv = bias[n];
#pragma unroll
    for (int v = 0; v < 8; ++v) {
      const int row = r0 + v + 8 * kg;
      const int tt = row & (T_SZ - 1), bb = row >> 9;
      // gx streams once (419MB > L2): keep it non-temporal in near caches
      __builtin_nontemporal_store(acc[nt][v] + bv,
          gx + (size_t)(tt * B_SZ + bb) * R4 + n);
    }
  }
}

// ---------------- persistent LSTM recurrence ------------------------------
// grid 20 (= 2 dirs * 10 chunks), block 256, dynamic LDS = 160*416*2 + 32*160*4
__global__ __launch_bounds__(256) void k_rnn(
    const bf16_t* __restrict__ Whh_l, const bf16_t* __restrict__ Whh_r,
    const float* __restrict__ gx_l, const float* __restrict__ gx_r,
    bf16_t* __restrict__ hbuf_l, bf16_t* __restrict__ hbuf_r,
    unsigned* __restrict__ ctr, float* __restrict__ out) {
  extern __shared__ char smem[];
  bf16_t* whh  = (bf16_t*)smem;                                  // [160][KP] gate-interleaved
  float*  glds = (float*)(smem + (size_t)NLOC * KP * sizeof(bf16_t)); // [32][160]

  const int dir   = blockIdx.x / NCHUNK;
  const int chunk = blockIdx.x % NCHUNK;
  const bf16_t* Whh = dir ? Whh_r : Whh_l;
  const float*  gx  = dir ? gx_r  : gx_l;
  bf16_t* hbuf = dir ? hbuf_r : hbuf_l;
  unsigned* cptr = ctr + dir;

  const int tid = threadIdx.x, w = tid >> 5, lane = tid & 31;
  const int mc = lane & 15, kg = lane >> 4;
  const int mt = w & 3, ns = w >> 2;           // wave tile: rows [mt*16,+16), cols [ns*80,+80)

  // ---- stage Whh slice into LDS with gate interleave (n_local = 4*r + g) ----
#ifdef USE_TDM
  if (tid < 32) {   // one TDM descriptor, issued by wave 0 (EXEC ignored by TDM)
    // 3-D tile: x = 416 bf16 (one row), y = 4 gates (stride 400 rows),
    // z = 40 hidden units (stride 1 row).  LDS fill order (z,y,x) == interleave.
    const bf16_t* gsrc = Whh + (size_t)(chunk * RCHUNK) * KP;
    unsigned long long ga = (unsigned long long)(const void*)gsrc;
    unsigned ldsoff = (unsigned)(unsigned long long)(uintptr_t)(void*)whh;
    v4u g0 = { 0x1u,                    // count = 1 valid descriptor
               ldsoff,                  // lds_addr (bytes)
               (unsigned)ga,            // global_addr[31:0]
               (unsigned)(ga >> 32) | 0x80000000u };   // addr[56:32] | type=2
    v8i_t g1 = { (int)0x00010000u,                      // data_size = 2B
                 (int)(416u << 16),                     // tensor_dim0[15:0]
                 (int)((416u >> 16) | (4u << 16)),      // dim0 hi | tensor_dim1 lo
                 (int)(416u << 16),                     // dim1 hi | tile_dim0 = 416
                 (int)(4u | (40u << 16)),               // tile_dim1 = 4, tile_dim2 = 40
                 (int)(400u * 416u),                    // tensor_dim0_stride lo
                 (int)(416u << 16),                     // stride0 hi | tensor_dim1_stride lo
                 0 };                                   // stride1 hi
    v4i_t g2 = { 40, 0, 0, 0 };                         // tensor_dim2 = 40
    v4i_t g3 = { 0, 0, 0, 0 };
#if __clang_major__ >= 23
    v8i_t g4 = { 0, 0, 0, 0, 0, 0, 0, 0 };
    __builtin_amdgcn_tensor_load_to_lds(g0, g1, g2, g3, g4, 0);
#else
    __builtin_amdgcn_tensor_load_to_lds(g0, g1, g2, g3, 0);
#endif
    __builtin_amdgcn_s_wait_tensorcnt(0);
  }
#else
  for (int e = tid; e < NLOC * (KP / 8); e += 256) {
    const int nl = e / (KP / 8);
    const int kk = (e - nl * (KP / 8)) * 8;
    const int g = nl & 3, rl = nl >> 2;
    const bf16_t* src = Whh + (size_t)(g * HID + chunk * RCHUNK + rl) * KP + kk;
    *(v8bf*)(whh + (size_t)nl * KP + kk) = *(const v8bf*)src;
  }
#endif
  __syncthreads();

  float creg[10];
#pragma unroll
  for (int i = 0; i < 10; ++i) creg[i] = 0.0f;

  for (int s = 0; s < T_SZ; ++s) {
    const int t = dir ? (T_SZ - 1 - s) : s;
    const bf16_t* hprev = hbuf + (size_t)(s & 1) * B_SZ * KP;
    bf16_t*       hnext = hbuf + (size_t)((s + 1) & 1) * B_SZ * KP;

    // prefetch next step's gx rows for this chunk (global_prefetch_b8)
    const int tn = dir ? (t > 0 ? t - 1 : t) : (t < T_SZ - 1 ? t + 1 : t);
    __builtin_prefetch(gx + (size_t)(tn * B_SZ + (tid & 63)) * R4
                          + (tid >> 6) * HID + chunk * RCHUNK, 0, 1);

    // hpart = h_prev @ Whh_slice^T   (M=64, K=416, N=160 interleaved)
    v8f acc[5];
#pragma unroll
    for (int i = 0; i < 5; ++i) acc[i] = {};
    const bf16_t* arow = hprev + (size_t)(mt * 16 + mc) * KP;
    for (int ks = 0; ks < 13; ++ks) {
      const int k0 = ks * 32;
      v16bf a = cat8(*(const v8bf*)(arow + k0 + 8 * kg),
                     *(const v8bf*)(arow + k0 + 16 + 8 * kg));
#pragma unroll
      for (int nt = 0; nt < 5; ++nt) {
        const bf16_t* bp = whh + (size_t)(ns * 80 + nt * 16 + mc) * KP + k0 + 8 * kg;
        acc[nt] = wmma_bf16(a, cat8(*(const v8bf*)bp, *(const v8bf*)(bp + 16)), acc[nt]);
      }
    }

    // epilogue in two 32-row phases through LDS
#pragma unroll
    for (int ph = 0; ph < 2; ++ph) {
      __syncthreads();
      if ((mt >> 1) == ph) {
#pragma unroll
        for (int nt = 0; nt < 5; ++nt) {
          const int nl = ns * 80 + nt * 16 + mc;
          const int mbase = (mt & 1) * 16 + 8 * kg;
#pragma unroll
          for (int v = 0; v < 8; ++v) glds[(mbase + v) * NLOC + nl] = acc[nt][v];
        }
      }
      __syncthreads();
#pragma unroll
      for (int j = 0; j < 5; ++j) {
        const int q = tid * 5 + j;                 // 1280 elems / phase
        const int ml = q / RCHUNK, rl = q - ml * RCHUNK;
        const int mrow = ph * 32 + ml;
        const int rg = chunk * RCHUNK + rl;
        const float* g4  = glds + ml * NLOC + rl * 4;
        const float* gxr = gx + (size_t)(t * B_SZ + mrow) * R4 + rg;
        const float gi = g4[0] + __builtin_nontemporal_load(gxr);            // last use
        const float gf = g4[1] + __builtin_nontemporal_load(gxr + HID);
        const float gg = g4[2] + __builtin_nontemporal_load(gxr + 2 * HID);
        const float go = g4[3] + __builtin_nontemporal_load(gxr + 3 * HID);
        const float c  = sigmoidf_fast(gf) * creg[ph * 5 + j]
                       + sigmoidf_fast(gi) * tanhf(gg);
        creg[ph * 5 + j] = c;
        const float h = sigmoidf_fast(go) * tanhf(c);
        __builtin_nontemporal_store(h,
            out + ((size_t)mrow * T_SZ + t) * 800 + dir * HID + rg);
        hnext[(size_t)mrow * KP + rg] = (bf16_t)h;
      }
    }
    __syncthreads();

    // device-scope barrier across the 10 WGs of this direction
    __threadfence();
    if (tid == 0) {
      __hip_atomic_fetch_add(cptr, 1u, __ATOMIC_RELEASE, __HIP_MEMORY_SCOPE_AGENT);
      const unsigned tgt = (unsigned)(s + 1) * NCHUNK;
      while (__hip_atomic_load(cptr, __ATOMIC_ACQUIRE, __HIP_MEMORY_SCOPE_AGENT) < tgt)
        __builtin_amdgcn_s_sleep(1);
    }
    __syncthreads();
    __threadfence();
  }
}

// ---------------- host ----------------
extern "C" void kernel_launch(void* const* d_in, const int* in_sizes, int n_in,
                              void* d_out, int out_size, void* d_ws, size_t ws_size,
                              hipStream_t stream) {
  const int*   charIdx    = (const int*)d_in[0];
  const int*   bicharIdx  = (const int*)d_in[1];
  const float* hidden     = (const float*)d_in[2];
  const float* extCharW   = (const float*)d_in[3];
  const float* extBiCharW = (const float*)d_in[4];
  const float* charW      = (const float*)d_in[5];
  const float* bicharW    = (const float*)d_in[6];
  const float* leftW      = (const float*)d_in[7];
  const float* leftb      = (const float*)d_in[8];
  const float* rightW     = (const float*)d_in[9];
  const float* rightb     = (const float*)d_in[10];
  const float* Wih_l      = (const float*)d_in[11];
  const float* Whh_l      = (const float*)d_in[12];
  const float* bih_l      = (const float*)d_in[13];
  const float* bhh_l      = (const float*)d_in[14];
  const float* Wih_r      = (const float*)d_in[15];
  const float* Whh_r      = (const float*)d_in[16];
  const float* bih_r      = (const float*)d_in[17];
  const float* bhh_r      = (const float*)d_in[18];
  float* out = (float*)d_out;
  char*  ws  = (char*)d_ws;

  bf16_t* lWbf   = (bf16_t*)(ws + O_LW);
  bf16_t* rWbf   = (bf16_t*)(ws + O_RW);
  bf16_t* WihLbf = (bf16_t*)(ws + O_WIHL);
  bf16_t* WihRbf = (bf16_t*)(ws + O_WIHR);
  bf16_t* WhhLbf = (bf16_t*)(ws + O_WHHL);
  bf16_t* WhhRbf = (bf16_t*)(ws + O_WHHR);
  float*  biasL  = (float*)(ws + O_BIASL);
  float*  biasR  = (float*)(ws + O_BIASR);
  bf16_t* hbufL  = (bf16_t*)(ws + O_HBUFL);
  bf16_t* hbufR  = (bf16_t*)(ws + O_HBUFR);
  unsigned* ctr  = (unsigned*)(ws + O_CTR);
  bf16_t* InL    = (bf16_t*)(ws + O_INL);
  bf16_t* InR    = (bf16_t*)(ws + O_INR);
  float*  gxL    = (float*)(ws + O_GXL);
  float*  gxR    = (float*)(ws + O_GXR);

  // --- prep: weight conversion, bias fusion, state zeroing ---
  k_cvt_dense<<<(400 * 800 + 255) / 256, 256, 0, stream>>>(leftW, lWbf, 400 * 800);
  k_cvt_dense<<<(400 * 800 + 255) / 256, 256, 0, stream>>>(rightW, rWbf, 400 * 800);
  k_cvt_pad416<<<(R4 * KP + 255) / 256, 256, 0, stream>>>(Wih_l, WihLbf);
  k_cvt_pad416<<<(R4 * KP + 255) / 256, 256, 0, stream>>>(Wih_r, WihRbf);
  k_cvt_pad416<<<(R4 * KP + 255) / 256, 256, 0, stream>>>(Whh_l, WhhLbf);
  k_cvt_pad416<<<(R4 * KP + 255) / 256, 256, 0, stream>>>(Whh_r, WhhRbf);
  k_bias_sum<<<(R4 + 255) / 256, 256, 0, stream>>>(bih_l, bhh_l, biasL);
  k_bias_sum<<<(R4 + 255) / 256, 256, 0, stream>>>(bih_r, bhh_r, biasR);
  k_zero_pad_cols<<<(M_TOT * 16 + 255) / 256, 256, 0, stream>>>(InL);
  k_zero_pad_cols<<<(M_TOT * 16 + 255) / 256, 256, 0, stream>>>(InR);
  {   // zero hbufL, hbufR, counters (contiguous region)
    const int nz = (int)((O_INL - O_HBUFL) / 4);
    k_zero_u32<<<(nz + 255) / 256, 256, 0, stream>>>((unsigned*)(ws + O_HBUFL), nz);
  }

  // --- phase 1: fused gather + projection (WMMA bf16) ---
  k_proj<<<dim3(M_TOT / 128, HID / 80, 2), 256, 0, stream>>>(
      charIdx, bicharIdx, charW, extCharW, bicharW, extBiCharW,
      lWbf, rWbf, leftb, rightb, InL, InR);

  // --- phase 2: x-side gate precompute (WMMA bf16) ---
  k_gates<<<dim3(M_TOT / 128, R4 / 80, 2), 256, 0, stream>>>(
      InL, InR, WihLbf, WihRbf, biasL, biasR, gxL, gxR);

  // --- phase 3: persistent bidirectional LSTM recurrence ---
  const size_t shmem = (size_t)NLOC * KP * sizeof(bf16_t) + 32 * NLOC * sizeof(float);
  k_rnn<<<2 * NCHUNK, 256, shmem, stream>>>(
      WhhLbf, WhhRbf, gxL, gxR, hbufL, hbufR, ctr, out);

  // --- pass-through hidden output ---
  hipMemcpyAsync(out + (size_t)B_SZ * T_SZ * 800, hidden,
                 (size_t)2 * B_SZ * HID * sizeof(float),
                 hipMemcpyDeviceToDevice, stream);
}
